// FeedForward_denoise_25151328485789
// MI455X (gfx1250) — matrix operand order repaired and checked by reading, not compiled, not run on hardware
//
#include <hip/hip_runtime.h>
#include <cstdint>

// ---------------------------------------------------------------------------
// Fused FeedForward (denoise FFN) for MI455X / gfx1250, wave32.
//   TDM-staged x tile -> in-LDS f16 transpose -> project_in (WMMA f16 GEMM)
//   grouped 3x3 convs (VALU from LDS) -> exact-GELU gate -> g in LDS
//   project_out (WMMA f16 GEMM, groups=2) -> global
// ---------------------------------------------------------------------------

#define DIMC 64
#define HID  128
#define XCCH 384
#define HH   256
#define WW_  256
#define HWSZ (HH * WW_)
#define TS   8
#define HALO 12      // 12x12 halo patch (halo of 2 each side)
#define NPX  144     // 12*12
#define NPX1 100     // 10*10 (d3a grid, halo of 1)

typedef __attribute__((ext_vector_type(16))) _Float16 v16h;
typedef __attribute__((ext_vector_type(8)))  float    v8f;
typedef __attribute__((ext_vector_type(4)))  unsigned int u32x4;
typedef __attribute__((ext_vector_type(8)))  int      i32x8;
typedef __attribute__((ext_vector_type(4)))  int      i32x4;

#ifndef __has_builtin
#define __has_builtin(x) 0
#endif
#if __has_builtin(__builtin_amdgcn_tensor_load_to_lds)
#define HAVE_TDM 1
#else
#define HAVE_TDM 0
#endif
#if __has_builtin(__builtin_amdgcn_s_wait_tensorcnt)
#define HAVE_TWAIT 1
#else
#define HAVE_TWAIT 0
#endif

// A-matrix (16-bit, 16x32) per-lane K pattern: VGPR r holds K={2r,2r+1} (r<4)
// or K={16+2(r-4), ...} (r>=4); +8 for lanes 16..31.  (CDNA5 ISA)
__device__ __forceinline__ int kpatA(int r, int lane) {
  return (r < 4 ? 2 * r : 16 + 2 * (r - 4)) + ((lane & 16) ? 8 : 0);
}
// B-matrix (16-bit, 32x16) per-lane K pattern: same as A: VGPR r holds
// K={2r,2r+1} (r<4) or K={16+2(r-4), ...} (r>=4); +8 for lanes 16..31.
__device__ __forceinline__ int kpatB(int r, int lane) {
  return (r < 4 ? 2 * r : 16 + 2 * (r - 4)) + ((lane & 16) ? 8 : 0);
}

__device__ __forceinline__ v8f wmma_f16(v16h a, v16h b, v8f c) {
  return __builtin_amdgcn_wmma_f32_16x16x32_f16(
      /*neg_a=*/false, a, /*neg_b=*/false, b,
      /*c_mod=*/(short)0, c, /*reuse_a=*/false, /*reuse_b=*/false);
}

// Build a B fragment (32x16 f16) from an f32 row-major weight matrix
// w[row = out channel][ldk], columns ko..ko+31.
__device__ __forceinline__ v16h build_B(const float* wrow, int ko, int lane) {
  v16h b;
#pragma unroll
  for (int r = 0; r < 8; ++r) {
    int kb = ko + kpatB(r, lane);
    b[2 * r]     = (_Float16)wrow[kb];
    b[2 * r + 1] = (_Float16)wrow[kb + 1];
  }
  return b;
}

// Build an A fragment (16x32 f16) from a pixel-major f16 row (64 or 128 ch).
__device__ __forceinline__ v16h build_A(const _Float16* arow, int ko, int lane) {
  v16h a;
#pragma unroll
  for (int r = 0; r < 8; ++r) {
    int ka = ko + kpatA(r, lane);
    a[2 * r]     = arow[ka];      // consecutive halves -> merged ds_load_b32
    a[2 * r + 1] = arow[ka + 1];
  }
  return a;
}

__device__ __forceinline__ float gelu_exact(float v) {
  return 0.5f * v * (1.0f + erff(v * 0.70710678118654752f));
}

// grouped 3x3 conv (3 in-ch/group) read from xc at halo-2 grid.
__device__ __forceinline__ float dw3x3_g3(const _Float16* xc, const float* w27,
                                          float bias, int oy, int ox, int ch) {
  float acc = bias;
  const float* wc = w27 + ch * 27;
#pragma unroll
  for (int dy = 0; dy < 3; ++dy)
#pragma unroll
    for (int dx = 0; dx < 3; ++dx) {
      const _Float16* xr = xc + ((oy + dy) * HALO + (ox + dx)) * XCCH + 3 * ch;
      int t = dy * 3 + dx;
      acc += (float)xr[0] * wc[t] + (float)xr[1] * wc[9 + t] +
             (float)xr[2] * wc[18 + t];
    }
  return acc;
}

__global__ __launch_bounds__(256) void ffn_fused(
    const float* __restrict__ x, const float* __restrict__ pne,
    const float* __restrict__ w_in, const float* __restrict__ b_in,
    const float* __restrict__ w_pn, const float* __restrict__ b_pn,
    const float* __restrict__ w_dw1, const float* __restrict__ b_dw1,
    const float* __restrict__ w_dw3a, const float* __restrict__ b_dw3a,
    const float* __restrict__ w_dw3b, const float* __restrict__ b_dw3b,
    const float* __restrict__ w_out, const float* __restrict__ b_out,
    float* __restrict__ out) {
  // LDS layout (152576 B total):
  //   [0      .. 36864)  stage:  x tile f32 [64][144]   (TDM destination)
  //   [0      .. 18432)  phase1: xh f16 [144][64]       (in-place transpose)
  //   [0      .. 25600)  phase2: d3a f16 [100][128]     (overlays dead xh)
  //   [25600  .. 41984)  phase2: g   f16 [64][128]
  //   [41984  .. 152576) xc f16 [144][384] (pn: ch 0..127, xi: ch 128..383)
  __shared__ __align__(16) unsigned char smem[152576];
  float*    xl  = (float*)smem;
  _Float16* xh  = (_Float16*)smem;
  _Float16* d3a = (_Float16*)smem;
  _Float16* gb  = (_Float16*)(smem + 25600);
  _Float16* xc  = (_Float16*)(smem + 41984);

  const int tid  = threadIdx.x;
  const int wave = tid >> 5;
  const int lane = tid & 31;
  const int M    = lane & 15;             // row/col within a 16-wide fragment
  const int Mb   = (lane & 16) ? 8 : 0;   // C/D: M = r + 8*(lane>=16)
  const int bz = blockIdx.z;
  const int h0 = blockIdx.y * TS, w0 = blockIdx.x * TS;
  int hh = h0 - 2; hh = hh < 0 ? 0 : (hh > HH - HALO ? HH - HALO : hh);
  int ww = w0 - 2; ww = ww < 0 ? 0 : (ww > WW_ - HALO ? WW_ - HALO : ww);

  if (tid == 0) {
    __builtin_prefetch(w_in, 0, 0);   // global_prefetch_b8: weights -> L2
    __builtin_prefetch(w_out, 0, 0);
  }

#if HAVE_TDM
  // ---- Stage x tile [64 ch][12][12] into LDS via the Tensor Data Mover ----
  if (wave == 0) {
    unsigned long long ga =
        (unsigned long long)(uintptr_t)x +
        4ull * ((unsigned long long)bz * DIMC * HWSZ +
                (unsigned long long)hh * WW_ + (unsigned long long)ww);
    unsigned ldsa = (unsigned)(uintptr_t)xl;
    u32x4 g0;
    g0[0] = 1u;                                   // count=1, user descriptor
    g0[1] = ldsa;                                 // lds_addr
    g0[2] = (unsigned)ga;                         // global_addr[31:0]
    g0[3] = (unsigned)((ga >> 32) & 0x1FFFFFFu)   // global_addr[56:32]
            | (2u << 30);                         // type=2 ("image")
    i32x8 g1;
    g1[0] = 0x20000;                              // data_size=2 (4B elements)
    g1[1] = (int)(256u << 16);                    // tensor_dim0 = W = 256
    g1[2] = (int)(256u << 16);                    // tensor_dim1 = H = 256
    g1[3] = (int)(12u << 16);                     // tile_dim0 = 12
    g1[4] = (int)(12u | (64u << 16));             // tile_dim1=12, tile_dim2=64
    g1[5] = 256;                                  // dim0 stride = W
    g1[6] = 0;                                    // dim1 stride lo16
    g1[7] = 1;                                    // dim1 stride = H*W = 65536
    i32x4 g2; g2[0] = 64; g2[1] = 0; g2[2] = 0; g2[3] = 0;  // tensor_dim2=64
    i32x4 g3; g3[0] = 0;  g3[1] = 0; g3[2] = 0; g3[3] = 0;
#if __clang_major__ >= 23
    i32x8 g4; g4[0]=0;g4[1]=0;g4[2]=0;g4[3]=0;g4[4]=0;g4[5]=0;g4[6]=0;g4[7]=0;
    __builtin_amdgcn_tensor_load_to_lds(g0, g1, g2, g3, g4, 0);
#else
    __builtin_amdgcn_tensor_load_to_lds(g0, g1, g2, g3, 0);
#endif
  }
#endif

  // ---- pn projection (2 -> 128) into xc channels 0..127 (independent of x)
  for (int idx = tid; idx < NPX * HID; idx += 256) {  // 18432/256 = 72, exact
    int p = idx >> 7, c = idx & 127;
    int gy = hh + p / HALO, gx = ww + p % HALO;
    float p0 = pne[((bz * 2 + 0) * HH + gy) * WW_ + gx];
    float p1 = pne[((bz * 2 + 1) * HH + gy) * WW_ + gx];
    xc[p * XCCH + c] =
        (_Float16)(w_pn[2 * c] * p0 + w_pn[2 * c + 1] * p1 + b_pn[c]);
  }

#if !HAVE_TDM
  // Fallback: cooperative global->LDS copy, same [c][p] layout as TDM tile.
  for (int idx = tid; idx < DIMC * NPX; idx += 256) {
    int c = idx / NPX, p = idx % NPX;
    xl[idx] = x[((unsigned long long)(bz * DIMC + c)) * HWSZ +
                (hh + p / HALO) * WW_ + (ww + p % HALO)];
  }
#endif
#if HAVE_TDM && HAVE_TWAIT
  if (wave == 0) __builtin_amdgcn_s_wait_tensorcnt(0);
#endif
  __syncthreads();

  // ---- Phase 0.5: in-place f32 [c][p] -> f16 pixel-major [p][64] transpose.
  // 144 px * 32 channel-pairs = 4608 pairs; 18 per thread, staged via regs.
  {
    float ra[18], rb[18];
#pragma unroll
    for (int k = 0; k < 18; ++k) {
      int idx = tid + k * 256;
      int p = idx >> 5, cp = idx & 31;
      ra[k] = xl[(2 * cp) * NPX + p];
      rb[k] = xl[(2 * cp + 1) * NPX + p];
    }
    __syncthreads();
#pragma unroll
    for (int k = 0; k < 18; ++k) {
      int idx = tid + k * 256;
      int p = idx >> 5, cp = idx & 31;
      union { _Float16 h[2]; unsigned u; } pk;
      pk.h[0] = (_Float16)ra[k];
      pk.h[1] = (_Float16)rb[k];
      ((unsigned*)xh)[p * 32 + cp] = pk.u;   // xh[p][2cp..2cp+1]
    }
    __syncthreads();
  }

  // ---- Phase 1: project_in (64 -> 256) as WMMA GEMM over 144 halo pixels.
  // Each wave owns out-tiles {wave, wave+8}: build B once, sweep 9 px-tiles.
  for (int rep = 0; rep < 2; ++rep) {
    int ot = wave + 8 * rep;                         // 0..15
    const float* wrow = w_in + (ot * 16 + M) * DIMC; // row = out channel N
    v16h b0 = build_B(wrow, 0, lane);
    v16h b1 = build_B(wrow, 32, lane);
    float bias = b_in[ot * 16 + M];
#pragma unroll
    for (int pt = 0; pt < 9; ++pt) {
      const _Float16* arow = xh + (pt * 16 + M) * DIMC;
      v8f acc = {0.f, 0.f, 0.f, 0.f, 0.f, 0.f, 0.f, 0.f};
      acc = wmma_f16(build_A(arow, 0, lane), b0, acc);
      acc = wmma_f16(build_A(arow, 32, lane), b1, acc);
      int o = ot * 16 + M;
#pragma unroll
      for (int r = 0; r < 8; ++r) {
        int dp = pt * 16 + Mb + r;
        xc[dp * XCCH + 128 + o] = (_Float16)(acc[r] + bias);
      }
    }
  }
  __syncthreads();

  // ---- Phase 2a: dwconvIII[0] (grouped 3x3, 3 in-ch/group) at 10x10 halo-1.
  for (int idx = tid; idx < NPX1 * HID; idx += 256) {  // 12800/256 = 50, exact
    int p3 = idx >> 7, c = idx & 127;
    int yy = p3 / 10, xx = p3 % 10;
    float acc = b_dw3a[c];
    const float* wc = w_dw3a + c * 27;
#pragma unroll
    for (int dy = 0; dy < 3; ++dy)
#pragma unroll
      for (int dx = 0; dx < 3; ++dx) {
        const _Float16* xr = xc + ((yy + dy) * HALO + (xx + dx)) * XCCH + 3 * c;
        int t = dy * 3 + dx;
        acc += (float)xr[0] * wc[t] + (float)xr[1] * wc[9 + t] +
               (float)xr[2] * wc[18 + t];
      }
    d3a[p3 * HID + c] = (_Float16)acc;
  }
  __syncthreads();

  // ---- Phase 2b: dwconvI + dwconvIII[1] + channel shuffle + exact GELU gate.
  for (int idx = tid; idx < 64 * HID; idx += 256) {  // 8192/256 = 32, exact
    int p = idx >> 7, c = idx & 127;
    int y = p >> 3, xp = p & 7;
    float x1v, x2v;
    if (c < 64) {
      // x1 = d1[c], x2 = d1[c+64]
      x1v = dw3x3_g3(xc, w_dw1, b_dw1[c], y + 1, xp + 1, c);
      x2v = dw3x3_g3(xc, w_dw1, b_dw1[c + 64], y + 1, xp + 1, c + 64);
    } else {
      // x1 = d3b[c-64], x2 = d3b[c]  (depthwise 3x3 over d3a)
      float a0 = b_dw3b[c - 64], a1 = b_dw3b[c];
      const float* w0c = w_dw3b + (c - 64) * 9;
      const float* w1c = w_dw3b + c * 9;
#pragma unroll
      for (int dy = 0; dy < 3; ++dy)
#pragma unroll
        for (int dx = 0; dx < 3; ++dx) {
          const _Float16* dr = d3a + ((y + dy) * 10 + (xp + dx)) * HID;
          int t = dy * 3 + dx;
          a0 += (float)dr[c - 64] * w0c[t];
          a1 += (float)dr[c] * w1c[t];
        }
      x1v = a0; x2v = a1;
    }
    gb[p * HID + c] = (_Float16)(gelu_exact(x1v) * x2v);
  }
  __syncthreads();

  // ---- Phase 3: project_out (128 -> 64, groups=2) as WMMA GEMM on 8x8 tile.
  // Wave -> fixed out-tile (ot = wave&3), B built once; two pixel tiles each.
  {
    int ot = wave & 3;
    int ptbase = wave >> 2;                     // 0 or 1
    int grp_base = (ot >= 2) ? 64 : 0;          // group 0/1 input channels
    const float* wrow = w_out + (ot * 16 + M) * 64;   // [o][64] within group
    v16h b0 = build_B(wrow, 0, lane);
    v16h b1 = build_B(wrow, 32, lane);
    int o = ot * 16 + M;
    float bias = b_out[o];
#pragma unroll
    for (int rep = 0; rep < 2; ++rep) {
      int pt = ptbase + 2 * rep;                // waves cover pt 0..3
      const _Float16* arow = gb + (pt * 16 + M) * HID + grp_base;
      v8f acc = {0.f, 0.f, 0.f, 0.f, 0.f, 0.f, 0.f, 0.f};
      acc = wmma_f16(build_A(arow, 0, lane), b0, acc);
      acc = wmma_f16(build_A(arow, 32, lane), b1, acc);
#pragma unroll
      for (int r = 0; r < 8; ++r) {
        int dp = pt * 16 + Mb + r;
        int y = dp >> 3, xp = dp & 7;
        out[((unsigned long long)(bz * DIMC + o)) * HWSZ +
            (unsigned long long)(h0 + y) * WW_ + (w0 + xp)] = acc[r] + bias;
      }
    }
  }
}

extern "C" void kernel_launch(void* const* d_in, const int* in_sizes, int n_in,
                              void* d_out, int out_size, void* d_ws,
                              size_t ws_size, hipStream_t stream) {
  (void)in_sizes; (void)n_in; (void)out_size; (void)d_ws; (void)ws_size;
  dim3 grid(WW_ / TS, HH / TS, 4);
  ffn_fused<<<grid, 256, 0, stream>>>(
      (const float*)d_in[0],  (const float*)d_in[1],  (const float*)d_in[2],
      (const float*)d_in[3],  (const float*)d_in[4],  (const float*)d_in[5],
      (const float*)d_in[6],  (const float*)d_in[7],  (const float*)d_in[8],
      (const float*)d_in[9],  (const float*)d_in[10], (const float*)d_in[11],
      (const float*)d_in[12], (const float*)d_in[13], (float*)d_out);
}
